// MultiHeadHCGAttention_16587163698037
// MI455X (gfx1250) — matrix operand order repaired and checked by reading, not compiled, not run on hardware
//
#include <hip/hip_runtime.h>

// ---------------- types ----------------
typedef __attribute__((ext_vector_type(16))) __bf16 v16bf;
typedef __attribute__((ext_vector_type(8)))  __bf16 v8bf;
typedef __attribute__((ext_vector_type(8)))  float  v8f;

// ---------------- problem constants ----------------
static constexpr int S_  = 2048;
static constexpr int B_  = 4;
static constexpr int H_  = 1024;
static constexpr int NH_ = 8;
static constexpr int DK_ = 128;
// fold 1/sqrt(DK) * log2(e) into the Q projection so attention can use exp2 directly
static constexpr float QSCALE = 0.08838834764831845f * 1.4426950408889634f;
static constexpr float NEG_INF_IN = -1.0e9f;

// ---------------- helpers ----------------
static __device__ __forceinline__ __bf16 f2bf(float f) {
  unsigned u = __builtin_bit_cast(unsigned, f);
  unsigned r = u + 0x7FFFu + ((u >> 16) & 1u);   // round-to-nearest-even
  unsigned short h = (unsigned short)(r >> 16);
  return __builtin_bit_cast(__bf16, h);
}

// packed f32x2 -> bf16x2 (v_cvt_pk_bf16_f32 when available)
static __device__ __forceinline__ unsigned pack2bf(float x, float y) {
#if __has_builtin(__builtin_amdgcn_cvt_pk_bf16_f32)
  typedef __attribute__((ext_vector_type(2))) __bf16 v2bf;
  v2bf r = __builtin_amdgcn_cvt_pk_bf16_f32(x, y);
  return __builtin_bit_cast(unsigned, r);
#else
  unsigned lo = (unsigned)__builtin_bit_cast(unsigned short, f2bf(x));
  unsigned hi = (unsigned)__builtin_bit_cast(unsigned short, f2bf(y));
  return lo | (hi << 16);
#endif
}

union V16u { v16bf v; v8bf h[2]; };

static __device__ __forceinline__ v8f vzero8() {
  v8f z;
#pragma unroll
  for (int i = 0; i < 8; ++i) z[i] = 0.0f;
  return z;
}

// A-fragment (16x32 bf16, MxK): lane half hh, row = ln.
static __device__ __forceinline__ v16bf load_afrag(const __bf16* base, int row, int ld, int hh) {
  V16u u;
  u.h[0] = *(const v8bf*)(base + (size_t)row * ld + hh * 8);
  u.h[1] = *(const v8bf*)(base + (size_t)row * ld + 16 + hh * 8);
  return u.v;
}

// B-fragment (32x16 bf16, KxN) read from a tile stored TRANSPOSED as [n][k].
static __device__ __forceinline__ v16bf load_bfrag_tr(const __bf16* base, int col, int ld, int hh) {
  V16u u;
  u.h[0] = *(const v8bf*)(base + (size_t)col * ld + hh * 8);
  u.h[1] = *(const v8bf*)(base + (size_t)col * ld + 16 + hh * 8);
  return u.v;
}

static __device__ __forceinline__ v8f wmma_bf16(v16bf a, v16bf b, v8f c) {
  return __builtin_amdgcn_wmma_f32_16x16x32_bf16(false, a, false, b, (short)0, c, false, false);
}

static __device__ __forceinline__ float redmax16(float v) {
  v = fmaxf(v, __shfl_xor(v, 1));
  v = fmaxf(v, __shfl_xor(v, 2));
  v = fmaxf(v, __shfl_xor(v, 4));
  v = fmaxf(v, __shfl_xor(v, 8));
  return v;
}

// One 32-deep K step of the 128x128 block tile: load all fragments, then
// issue the 8 WMMAs back-to-back so the XDL pipe stays fed.
static __device__ __forceinline__ void gemm_step(const __bf16* Asl, const __bf16* Bsl,
                                                 v8f acc[2][4], int wr, int wc, int hh, int ln) {
  v16bf a0 = load_afrag(Asl, wr * 32 + ln, 40, hh);
  v16bf a1 = load_afrag(Asl, wr * 32 + 16 + ln, 40, hh);
  v16bf b0 = load_bfrag_tr(Bsl, wc * 64 + 0 * 16 + ln, 40, hh);
  v16bf b1 = load_bfrag_tr(Bsl, wc * 64 + 1 * 16 + ln, 40, hh);
  v16bf b2 = load_bfrag_tr(Bsl, wc * 64 + 2 * 16 + ln, 40, hh);
  v16bf b3 = load_bfrag_tr(Bsl, wc * 64 + 3 * 16 + ln, 40, hh);
  acc[0][0] = wmma_bf16(a0, b0, acc[0][0]);
  acc[1][0] = wmma_bf16(a1, b0, acc[1][0]);
  acc[0][1] = wmma_bf16(a0, b1, acc[0][1]);
  acc[1][1] = wmma_bf16(a1, b1, acc[1][1]);
  acc[0][2] = wmma_bf16(a0, b2, acc[0][2]);
  acc[1][2] = wmma_bf16(a1, b2, acc[1][2]);
  acc[0][3] = wmma_bf16(a0, b3, acc[0][3]);
  acc[1][3] = wmma_bf16(a1, b3, acc[1][3]);
}

// ==================================================================
// Kernel 1: per-head input projection  (q|k|v) @ W[n] + b[n] -> bf16
// grid = (S/128, NH, B), block = 256 (8 waves). Double-buffered LDS.
// ==================================================================
template <bool USE_MASK>
__global__ __launch_bounds__(256) void proj_kernel(const float* __restrict__ in,
                                                   const unsigned char* __restrict__ mask,
                                                   const float* __restrict__ W,
                                                   const float* __restrict__ bias,
                                                   float outscale,
                                                   __bf16* __restrict__ out) {
  __shared__ __align__(16) __bf16 Asl[2][128 * 40];
  __shared__ __align__(16) __bf16 Bsl[2][128 * 40];

  const int tid = threadIdx.x;
  const int w = tid >> 5, lane = tid & 31, hh = lane >> 4, ln = lane & 15;
  const int wr = w >> 1, wc = w & 1;
  const int m0 = blockIdx.x * 128;
  const int n  = blockIdx.y;
  const int b  = blockIdx.z;

  const float* Ain = in + (size_t)b * H_;        // row stride B_*H_
  const float* Wh  = W + (size_t)n * H_ * DK_;   // [H, DK] row-major
  const float* bh  = bias + (size_t)n * DK_;
  __bf16* Oout = out + (size_t)(b * NH_ + n) * S_ * DK_;

  v8f acc[2][4];
#pragma unroll
  for (int i = 0; i < 2; ++i)
#pragma unroll
    for (int j = 0; j < 4; ++j) acc[i][j] = vzero8();

  const int ar = tid >> 1, ak = (tid & 1) * 16;  // A stage: 128 rows x 32 k
  const int bk = tid >> 3, bn = (tid & 7) * 16;  // B stage: 32 k x 128 n
  bool mrow = false;
  if (USE_MASK) mrow = mask[(size_t)b * S_ + (m0 + ar)] != 0;

  auto stage = [&](int pb, int k0) {
    const float4* ap = (const float4*)(Ain + (size_t)(m0 + ar) * (B_ * H_) + k0 + ak);
#pragma unroll
    for (int i = 0; i < 4; ++i) {
      float4 f = ap[i];
      if (USE_MASK && mrow) { f.x = NEG_INF_IN; f.y = NEG_INF_IN; f.z = NEG_INF_IN; f.w = NEG_INF_IN; }
      unsigned* d = (unsigned*)(Asl[pb] + (size_t)ar * 40 + ak + i * 4);
      d[0] = pack2bf(f.x, f.y);
      d[1] = pack2bf(f.z, f.w);
    }
    const float4* wp = (const float4*)(Wh + (size_t)(k0 + bk) * DK_ + bn);
#pragma unroll
    for (int i = 0; i < 4; ++i) {
      float4 f = wp[i];
      Bsl[pb][(size_t)(bn + i * 4 + 0) * 40 + bk] = f2bf(f.x);
      Bsl[pb][(size_t)(bn + i * 4 + 1) * 40 + bk] = f2bf(f.y);
      Bsl[pb][(size_t)(bn + i * 4 + 2) * 40 + bk] = f2bf(f.z);
      Bsl[pb][(size_t)(bn + i * 4 + 3) * 40 + bk] = f2bf(f.w);
    }
  };

  stage(0, 0);
  __syncthreads();
  int p = 0;
  for (int k0 = 0; k0 < H_; k0 += 32) {
    if (k0 + 32 < H_) stage(p ^ 1, k0 + 32);
    gemm_step(Asl[p], Bsl[p], acc, wr, wc, hh, ln);
    __syncthreads();
    p ^= 1;
  }

#pragma unroll
  for (int tr = 0; tr < 2; ++tr) {
#pragma unroll
    for (int t = 0; t < 4; ++t) {
      const int col = wc * 64 + t * 16 + ln;
      const float bv = bh[col];
#pragma unroll
      for (int j = 0; j < 8; ++j) {
        const int row = m0 + wr * 32 + tr * 16 + hh * 8 + j;
        Oout[(size_t)row * DK_ + col] = f2bf((acc[tr][t][j] + bv) * outscale);
      }
    }
  }
}

// ==================================================================
// Kernel 2: flash attention. grid = (S/128, NH, B), block = 256.
// Wave-private 16-row query tile; block-shared double-buffered 32-key
// K/V tiles. Softmax denominator computed as an extra WMMA (P @ ones)
// instead of lane shuffles; only row-max uses the 4-step butterfly.
// Q already carries 1/sqrt(dk)*log2(e) so exp2 is used directly.
// ==================================================================
__global__ __launch_bounds__(256) void attn_kernel(const __bf16* __restrict__ Q,
                                                   const __bf16* __restrict__ K,
                                                   const __bf16* __restrict__ V,
                                                   __bf16* __restrict__ O) {
  __shared__ __align__(16) __bf16 Kl[2][32 * 136];    // K[key][d]
  __shared__ __align__(16) __bf16 Vt[2][128 * 40];    // V transposed [d][key]
  __shared__ __align__(16) __bf16 Pl[8 * 16 * 40];    // per-wave P scratch [r][k]

  const int tid = threadIdx.x;
  const int w = tid >> 5, lane = tid & 31, hh = lane >> 4, ln = lane & 15;
  const int b = blockIdx.z, n = blockIdx.y;
  const int q0 = blockIdx.x * 128 + w * 16;

  const size_t headoff = (size_t)(b * NH_ + n) * S_ * DK_;
  const __bf16* Qg = Q + headoff + (size_t)q0 * DK_;

  v16bf aQ[4];
#pragma unroll
  for (int i = 0; i < 4; ++i) aQ[i] = load_afrag(Qg + i * 32, ln, DK_, hh);

  // all-ones B fragment for the row-sum WMMA
  const __bf16 one_bf = __builtin_bit_cast(__bf16, (unsigned short)0x3F80);
  v16bf bOne;
#pragma unroll
  for (int i = 0; i < 16; ++i) bOne[i] = one_bf;

  v8f acc[8];
#pragma unroll
  for (int g = 0; g < 8; ++g) acc[g] = vzero8();
  v8f acc_l = vzero8();           // running softmax denominators (per row)
  float m_r[8];
#pragma unroll
  for (int j = 0; j < 8; ++j) m_r[j] = -1.0e30f;

  const int skey = tid >> 3;       // staging: key 0..31
  const int sd   = (tid & 7) * 16; // staging: d 0..112
  __bf16* Pw = Pl + w * (16 * 40);

  auto stage = [&](int pb, int kt) {
    const __bf16* Kg = K + headoff + (size_t)(kt * 32) * DK_;
    const __bf16* Vg = V + headoff + (size_t)(kt * 32) * DK_;
    v8bf k0v = *(const v8bf*)(Kg + (size_t)skey * DK_ + sd);
    v8bf k1v = *(const v8bf*)(Kg + (size_t)skey * DK_ + sd + 8);
    *(v8bf*)(Kl[pb] + (size_t)skey * 136 + sd)     = k0v;
    *(v8bf*)(Kl[pb] + (size_t)skey * 136 + sd + 8) = k1v;
    v8bf v0 = *(const v8bf*)(Vg + (size_t)skey * DK_ + sd);
    v8bf v1 = *(const v8bf*)(Vg + (size_t)skey * DK_ + sd + 8);
#pragma unroll
    for (int i = 0; i < 8; ++i) {
      Vt[pb][(size_t)(sd + i) * 40 + skey]     = v0[i];
      Vt[pb][(size_t)(sd + 8 + i) * 40 + skey] = v1[i];
    }
  };

  stage(0, 0);
  __syncthreads();
  int p = 0;

  for (int kt = 0; kt < S_ / 32; ++kt) {
    if (kt + 1 < S_ / 32) stage(p ^ 1, kt + 1);

    // ---- scores: 16 q-rows x 32 keys (software-pipelined B frags) ----
    v8f s0 = vzero8(), s1 = vzero8();
    v16bf kb0 = load_bfrag_tr(Kl[p] + 0, ln, 136, hh);
    v16bf kb1 = load_bfrag_tr(Kl[p] + 0, 16 + ln, 136, hh);
#pragma unroll
    for (int i = 0; i < 4; ++i) {
      v16bf nb0, nb1;
      if (i < 3) {
        nb0 = load_bfrag_tr(Kl[p] + (i + 1) * 32, ln, 136, hh);
        nb1 = load_bfrag_tr(Kl[p] + (i + 1) * 32, 16 + ln, 136, hh);
      }
      s0 = wmma_bf16(aQ[i], kb0, s0);
      s1 = wmma_bf16(aQ[i], kb1, s1);
      kb0 = nb0; kb1 = nb1;
    }

    // ---- online softmax (row max via butterfly; row sum via WMMA below) ----
#pragma unroll
    for (int j = 0; j < 8; ++j) {
      const float a0 = s0[j];   // already scaled by 1/sqrt(dk)*log2e
      const float a1 = s1[j];
      const float mx = redmax16(fmaxf(a0, a1));
      const float mnew = fmaxf(m_r[j], mx);
      const float alpha = exp2f(m_r[j] - mnew);
      m_r[j] = mnew;
      const float p0 = exp2f(a0 - mnew);
      const float p1 = exp2f(a1 - mnew);
      acc_l[j] *= alpha;
#pragma unroll
      for (int g = 0; g < 8; ++g) acc[g][j] *= alpha;
      Pw[(size_t)(hh * 8 + j) * 40 + ln]      = f2bf(p0);
      Pw[(size_t)(hh * 8 + j) * 40 + 16 + ln] = f2bf(p1);
    }
    __builtin_amdgcn_wave_barrier();  // intra-wave LDS ops are in-order

    // ---- PV (+ row-sum): P(16x32) @ [V | 1](32x(128+16)) ----
    v16bf aP = load_afrag(Pw, ln, 40, hh);
    acc_l = wmma_bf16(aP, bOne, acc_l);
    v16bf bV = load_bfrag_tr(Vt[p], ln, 40, hh);
#pragma unroll
    for (int g = 0; g < 8; ++g) {
      v16bf nbV;
      if (g < 7) nbV = load_bfrag_tr(Vt[p], (g + 1) * 16 + ln, 40, hh);
      acc[g] = wmma_bf16(aP, bV, acc[g]);
      bV = nbV;
    }
    __syncthreads();
    p ^= 1;
  }

  // ---- normalize and store O as [s][b][n*DK + d] bf16 ----
#pragma unroll
  for (int j = 0; j < 8; ++j) {
    const float inv = 1.0f / acc_l[j];
    const size_t orow = ((size_t)(q0 + hh * 8 + j) * B_ + b) * (NH_ * DK_) + (size_t)n * DK_;
#pragma unroll
    for (int g = 0; g < 8; ++g) O[orow + g * 16 + ln] = f2bf(acc[g][j] * inv);
  }
}

// ==================================================================
// Kernel 3: output projection  [8192 x 1024] @ Wo[1024 x 1024] + bo
// grid = (64, 8), block = 256. Double-buffered LDS, f32 out to d_out.
// ==================================================================
__global__ __launch_bounds__(256) void outproj_kernel(const __bf16* __restrict__ A,
                                                      const float* __restrict__ W,
                                                      const float* __restrict__ bias,
                                                      float* __restrict__ out) {
  __shared__ __align__(16) __bf16 Asl[2][128 * 40];
  __shared__ __align__(16) __bf16 Bsl[2][128 * 40];

  const int tid = threadIdx.x;
  const int w = tid >> 5, lane = tid & 31, hh = lane >> 4, ln = lane & 15;
  const int wr = w >> 1, wc = w & 1;
  const int m0 = blockIdx.x * 128;
  const int n0 = blockIdx.y * 128;
  const int KD = NH_ * DK_;

  v8f acc[2][4];
#pragma unroll
  for (int i = 0; i < 2; ++i)
#pragma unroll
    for (int j = 0; j < 4; ++j) acc[i][j] = vzero8();

  const int ar = tid >> 1, ak = (tid & 1) * 16;
  const int bk = tid >> 3, bn = (tid & 7) * 16;

  auto stage = [&](int pb, int k0) {
    v8bf a0 = *(const v8bf*)(A + (size_t)(m0 + ar) * KD + k0 + ak);
    v8bf a1 = *(const v8bf*)(A + (size_t)(m0 + ar) * KD + k0 + ak + 8);
    *(v8bf*)(Asl[pb] + (size_t)ar * 40 + ak)     = a0;
    *(v8bf*)(Asl[pb] + (size_t)ar * 40 + ak + 8) = a1;
    const float4* wp = (const float4*)(W + (size_t)(k0 + bk) * H_ + n0 + bn);
#pragma unroll
    for (int i = 0; i < 4; ++i) {
      float4 f = wp[i];
      Bsl[pb][(size_t)(bn + i * 4 + 0) * 40 + bk] = f2bf(f.x);
      Bsl[pb][(size_t)(bn + i * 4 + 1) * 40 + bk] = f2bf(f.y);
      Bsl[pb][(size_t)(bn + i * 4 + 2) * 40 + bk] = f2bf(f.z);
      Bsl[pb][(size_t)(bn + i * 4 + 3) * 40 + bk] = f2bf(f.w);
    }
  };

  stage(0, 0);
  __syncthreads();
  int p = 0;
  for (int k0 = 0; k0 < KD; k0 += 32) {
    if (k0 + 32 < KD) stage(p ^ 1, k0 + 32);
    gemm_step(Asl[p], Bsl[p], acc, wr, wc, hh, ln);
    __syncthreads();
    p ^= 1;
  }

#pragma unroll
  for (int tr = 0; tr < 2; ++tr) {
#pragma unroll
    for (int t = 0; t < 4; ++t) {
      const int col = n0 + wc * 64 + t * 16 + ln;
      const float bv = bias[col];
#pragma unroll
      for (int j = 0; j < 8; ++j) {
        const int row = m0 + wr * 32 + tr * 16 + hh * 8 + j;
        out[(size_t)row * H_ + col] = acc[tr][t][j] + bv;
      }
    }
  }
}

// ==================================================================
// launcher
// ==================================================================
extern "C" void kernel_launch(void* const* d_in, const int* in_sizes, int n_in,
                              void* d_out, int out_size, void* d_ws, size_t ws_size,
                              hipStream_t stream) {
  (void)in_sizes; (void)n_in; (void)out_size; (void)ws_size;

  const float* q  = (const float*)d_in[0];
  const float* k  = (const float*)d_in[1];
  const float* v  = (const float*)d_in[2];
  const unsigned char* mask = (const unsigned char*)d_in[3];
  const float* Wq = (const float*)d_in[4];
  const float* bq = (const float*)d_in[5];
  const float* Wk = (const float*)d_in[6];
  const float* bk = (const float*)d_in[7];
  const float* Wv = (const float*)d_in[8];
  const float* bv = (const float*)d_in[9];
  const float* Wo = (const float*)d_in[10];
  const float* bo = (const float*)d_in[11];
  float* out = (float*)d_out;

  const size_t n_elem = (size_t)B_ * NH_ * S_ * DK_;  // 8,388,608
  __bf16* Qb = (__bf16*)d_ws;
  __bf16* Kb = Qb + n_elem;
  __bf16* Vb = Kb + n_elem;
  __bf16* Ob = Vb + n_elem;

  dim3 blk(256);
  dim3 gproj(S_ / 128, NH_, B_);
  proj_kernel<false><<<gproj, blk, 0, stream>>>(q, nullptr, Wq, bq, QSCALE, Qb);
  proj_kernel<true> <<<gproj, blk, 0, stream>>>(k, mask,    Wk, bk, 1.0f,   Kb);
  proj_kernel<false><<<gproj, blk, 0, stream>>>(v, nullptr, Wv, bv, 1.0f,   Vb);

  attn_kernel<<<dim3(S_ / 128, NH_, B_), blk, 0, stream>>>(Qb, Kb, Vb, Ob);

  outproj_kernel<<<dim3((S_ * B_) / 128, H_ / 128, 1), blk, 0, stream>>>(Ob, Wo, bo, out);
}
